// ConvSTFT_12189117186579
// MI455X (gfx1250) — compile-verified, matrix operand order checked by reading, not run on hardware
//
#include <hip/hip_runtime.h>

typedef __attribute__((ext_vector_type(16))) _Float16 v16h;
typedef __attribute__((ext_vector_type(8)))  float    v8f;
typedef __attribute__((ext_vector_type(4)))  unsigned v4u;

#define T_OUT   963
#define F_OUT   257
#define NC      64
#define K_LEN   400
#define L_IN    96000
#define PAD     300
#define STRIDEH 100
#define KP      416            // K padded to 13 * 32
#define KSTEPS  13
#define MTILES  17             // ceil(257/16) frequency tiles
#define NT      4              // 4 n-tiles (64 t-columns) per workgroup
#define NCOLS   (NT * 16)
#define LDS_STR 424            // padded LDS row stride (halves) -> conflict-free
#define LDS_STRD (LDS_STR / 2) // row stride in dwords (212, multiple of 4)
#define BLOCK   288            // 9 waves -> at most 2 m-tile passes per wave
#define NWAVES  9
#define EPS_F   1.1920928955078125e-07f

#define FRAG_BYTES  1024                            // 32 lanes * 32 B
#define KSTEP_BYTES (4 * FRAG_BYTES)                // ahr | alr | ahi | ali
#define WPACK_BYTES ((long)MTILES * KSTEPS * KSTEP_BYTES)   // 905,216 B

union FragH { v16h h; v4u q[2]; unsigned u[8]; };

// split a float pair into hi-f16 pair and lo-f16 residual pair, packed as dwords
__device__ __forceinline__ void split_pack(float x, float y, unsigned& hi, unsigned& lo) {
    _Float16 hx = (_Float16)x, hy = (_Float16)y;
    _Float16 lx = (_Float16)(x - (float)hx), ly = (_Float16)(y - (float)hy);
    union { _Float16 h[2]; unsigned u; } a, b;
    a.h[0] = hx; a.h[1] = hy; hi = a.u;
    b.h[0] = lx; b.h[1] = ly; lo = b.u;
}

// ---- one-time weight pre-pack into WMMA A-fragment order (hi/lo, re/im) ----
__global__ __launch_bounds__(256) void pack_weight_kernel(
        const float* __restrict__ w, unsigned* __restrict__ wp) {
    int id = blockIdx.x * 256 + threadIdx.x;
    const int total = MTILES * KSTEPS * 4 * 32 * 8;
    if (id >= total) return;
    int j    = id & 7;
    int lane = (id >> 3) & 31;
    int ft   = (id >> 8) & 3;          // 0:ahr 1:alr 2:ahi 3:ali
    int rest = id >> 10;               // mt * KSTEPS + ks
    int ks = rest % KSTEPS;
    int mt = rest / KSTEPS;
    int g = lane >> 4, n16 = lane & 15;
    int row = mt * 16 + n16;
    int k = ks * 32 + g * 8 + (j < 4 ? 2 * j : 2 * j + 8);   // ISA 16-bit A K-map
    float x0 = 0.f, x1 = 0.f;
    if (row < F_OUT && k < K_LEN) {
        long r = row + ((ft & 2) ? F_OUT : 0);
        x0 = w[r * K_LEN + k];
        x1 = w[r * K_LEN + k + 1];
    }
    unsigned hi, lo;
    split_pack(x0, x1, hi, lo);
    wp[(long)rest * (KSTEP_BYTES / 4) + ft * (FRAG_BYTES / 4) + lane * 8 + j] =
        (ft & 1) ? lo : hi;
}

__global__ __launch_bounds__(BLOCK) void stft_wmma_kernel(
        const float* __restrict__ x, const unsigned* __restrict__ wp,
        float* __restrict__ out) {
    extern __shared__ _Float16 smem[];                  // 2 * NCOLS * LDS_STR halves
    _Float16* sHiH = smem;
    _Float16* sLoH = smem + NCOLS * LDS_STR;

    const int bc = blockIdx.y;
    const int t0 = blockIdx.x * NCOLS;
    const float* xin = x + (long)bc * L_IN;

    // ---- stage frame tile into LDS as hi/lo f16 ----
    for (int idx = threadIdx.x; idx < NCOLS * KP; idx += BLOCK) {
        int n = idx / KP, k = idx - n * KP;
        int t = t0 + n;
        int si = t * STRIDEH + k - PAD;
        float v = 0.f;
        if (k < K_LEN && t < T_OUT && si >= 0 && si < L_IN) v = xin[si];
        _Float16 h = (_Float16)v;
        _Float16 l = (_Float16)(v - (float)h);
        sHiH[n * LDS_STR + k] = h;
        sLoH[n * LDS_STR + k] = l;
    }
    __syncthreads();

    const int lane = threadIdx.x & 31;
    const int wave = threadIdx.x >> 5;
    const int g    = lane >> 4;
    const int n16  = lane & 15;
    const unsigned* sHi = (const unsigned*)sHiH;
    const unsigned* sLo = (const unsigned*)sLoH;

    const long mags_base  = (long)bc * F_OUT * T_OUT;
    const long phase_base = (long)NC * F_OUT * T_OUT;

    #pragma unroll 1
    for (int mt = wave; mt < MTILES; mt += NWAVES) {
        const int f0 = mt * 16;
        v8f accr[NT], acci[NT];
        #pragma unroll
        for (int nt = 0; nt < NT; ++nt) { accr[nt] = (v8f){}; acci[nt] = (v8f){}; }

        const char* pw0 = (const char*)wp + (long)mt * KSTEPS * KSTEP_BYTES + lane * 32;

        // keep rolled: full unroll (13x) explodes VGPR pressure into scratch spills
        #pragma unroll 1
        for (int ks = 0; ks < KSTEPS; ++ks) {
            // ---- A fragments: 8 coalesced global_load_b128, no guards ----
            const char* pw = pw0 + ks * KSTEP_BYTES;
            FragH ahr = *(const FragH*)(pw + 0 * FRAG_BYTES);
            FragH alr = *(const FragH*)(pw + 1 * FRAG_BYTES);
            FragH ahi = *(const FragH*)(pw + 2 * FRAG_BYTES);
            FragH ali = *(const FragH*)(pw + 3 * FRAG_BYTES);

            #pragma unroll
            for (int nt = 0; nt < NT; ++nt) {
                // 16B-aligned dword base: n*212 + ks*16 + g*4 (all multiples of 4)
                const int rb = (nt * 16 + n16) * LDS_STRD + ks * 16 + g * 4;
                const v4u* ph = (const v4u*)(sHi + rb);
                const v4u* pl = (const v4u*)(sLo + rb);
                FragH bh, bl;
                bh.q[0] = ph[0]; bh.q[1] = ph[2];   // dwords 0-3 and 8-11
                bl.q[0] = pl[0]; bl.q[1] = pl[2];

                accr[nt] = __builtin_amdgcn_wmma_f32_16x16x32_f16(false, ahr.h, false, bh.h, (short)0, accr[nt], false, false);
                accr[nt] = __builtin_amdgcn_wmma_f32_16x16x32_f16(false, ahr.h, false, bl.h, (short)0, accr[nt], false, false);
                accr[nt] = __builtin_amdgcn_wmma_f32_16x16x32_f16(false, alr.h, false, bh.h, (short)0, accr[nt], false, false);
                acci[nt] = __builtin_amdgcn_wmma_f32_16x16x32_f16(false, ahi.h, false, bh.h, (short)0, acci[nt], false, false);
                acci[nt] = __builtin_amdgcn_wmma_f32_16x16x32_f16(false, ahi.h, false, bl.h, (short)0, acci[nt], false, false);
                acci[nt] = __builtin_amdgcn_wmma_f32_16x16x32_f16(false, ali.h, false, bh.h, (short)0, acci[nt], false, false);
            }
        }

        // ---- fused epilogue: mag/phase straight from accumulators ----
        #pragma unroll
        for (int v = 0; v < 8; ++v) {
            int f = f0 + v + 8 * g;                    // C/D layout: M = v + 8*g
            #pragma unroll
            for (int nt = 0; nt < NT; ++nt) {
                int t = t0 + nt * 16 + n16;            // N = lane%16
                if (f < F_OUT && t < T_OUT) {
                    float r = accr[nt][v];
                    float i = acci[nt][v];
                    float m2 = r * r + i * i;
                    float m  = sqrtf(m2 < EPS_F ? EPS_F : m2);
                    float ph = atan2f(i + EPS_F, r + EPS_F);
                    long  o  = mags_base + (long)f * T_OUT + t;
                    out[o]              = m;
                    out[phase_base + o] = ph;
                }
            }
        }
    }
}

#define T_VAD 961
__global__ __launch_bounds__(256) void vad_kernel_g(
        const float* __restrict__ azi, int* __restrict__ vad) {
    int idx = blockIdx.x * blockDim.x + threadIdx.x;
    if (idx >= 16 * T_VAD) return;
    int p = idx / T_VAD, t = idx - p * T_VAD;
    const float* a = azi + (long)p * L_IN;
    int s0 = t * STRIDEH - PAD;
    float s = 0.f;
    for (int k = 0; k < 512; ++k) {
        int si = s0 + k;
        if (si >= 0 && si < L_IN) s += a[si];
    }
    vad[idx] = ((s * (1.0f / 512.0f)) >= (2.0f / 3.0f)) ? 1 : 0;
}

extern "C" void kernel_launch(void* const* d_in, const int* in_sizes, int n_in,
                              void* d_out, int out_size, void* d_ws, size_t ws_size,
                              hipStream_t stream) {
    const float* inputs = (const float*)d_in[0];   // (8,8,96000) f32
    const float* azi    = (const float*)d_in[1];   // (8,2,96000) f32
    const float* weight = (const float*)d_in[2];   // (514,1,400) f32
    float* out = (float*)d_out;                    // mags | phase | vad(int32)

    const int shmem = 2 * NCOLS * LDS_STR * (int)sizeof(_Float16);   // 108,544 B
    dim3 grid((T_OUT + NCOLS - 1) / NCOLS, NC);                      // (16, 64)

    hipFuncSetAttribute(reinterpret_cast<const void*>(stft_wmma_kernel),
                        hipFuncAttributeMaxDynamicSharedMemorySize, shmem);

    // pre-pack weight into WMMA fragment order (0.9 MB of workspace)
    const int npack = MTILES * KSTEPS * 4 * 32 * 8;
    pack_weight_kernel<<<(npack + 255) / 256, 256, 0, stream>>>(
        weight, (unsigned*)d_ws);

    stft_wmma_kernel<<<grid, BLOCK, shmem, stream>>>(
        inputs, (const unsigned*)d_ws, out);

    int* vad = (int*)out + 2L * NC * F_OUT * T_OUT;
    int nv = 16 * T_VAD;
    vad_kernel_g<<<(nv + 255) / 256, 256, 0, stream>>>(azi, vad);
}
